// MixerModel_25666724561037
// MI455X (gfx1250) — matrix operand order repaired and checked by reading, not compiled, not run on hardware
//
#include <hip/hip_runtime.h>
#include <hip/hip_bf16.h>
#include <math.h>

// ---------------- model constants (from reference) ----------------
constexpr int D      = 1024;     // d_model
constexpr int NL     = 4;        // layers per direction
constexpr int DIN    = 2048;     // d_inner
constexpr int S      = 16;       // d_state
constexpr int KC     = 4;        // d_conv
constexpr int R      = 64;       // dt_rank
constexpr int BATCH  = 2;
constexpr int L      = 2048;
constexpr int NTOK   = BATCH * L;          // 4096 rows
constexpr int XDBLW  = R + 2 * S;          // 96
constexpr int CH     = 64;                 // scan chunks
constexpr int CLEN   = L / CH;             // 32 steps per chunk

typedef __bf16 bf16_t;
typedef __attribute__((ext_vector_type(16))) __bf16 v16bf;
typedef __attribute__((ext_vector_type(8)))  float  v8f;
typedef unsigned int u32x4 __attribute__((ext_vector_type(4)));
typedef int          i32x4 __attribute__((ext_vector_type(4)));
typedef int          i32x8 __attribute__((ext_vector_type(8)));

// TDM availability (probe: exists on ROCm7.2 clang-22 [5 args] and clang-23 [6 args])
#if defined(__has_builtin)
#  if __has_builtin(__builtin_amdgcn_tensor_load_to_lds) && \
      __has_builtin(__builtin_amdgcn_s_wait_tensorcnt)
#    define HAVE_TDM 1
#  endif
#endif
#ifndef HAVE_TDM
#  define HAVE_TDM 0
#endif

// ---------------- helpers ----------------
__device__ __forceinline__ float siluf(float x) {
  return x / (1.f + __expf(-x));
}
__device__ __forceinline__ float softplusf(float x) {
  return (x > 20.f) ? x : log1pf(__expf(x));
}
__device__ __forceinline__ float block_reduce_sum(float v, float* red, int tid) {
  red[tid] = v;
  __syncthreads();
#pragma unroll
  for (int s = 128; s > 0; s >>= 1) {
    if (tid < s) red[tid] += red[tid + s];
    __syncthreads();
  }
  float r = red[0];
  __syncthreads();
  return r;
}

#if HAVE_TDM
// ---------------- Tensor Data Mover: 2-D bf16 tile -> LDS ----------------
// D# per CDNA5 ISA. Tile rows of BK=32 bf16 (64 B) are
// written with pad_interval=64B / pad_amount=8B so the LDS image matches the
// bank-padded [rows][BK+4] layout used by the WMMA fragment loads.
__device__ __forceinline__ void tdm_load_tile2d(
    unsigned lds_addr, const bf16_t* gsrc,
    unsigned tile_k, unsigned tile_rows,
    unsigned tensor_k, unsigned tensor_rows, unsigned stride_elems) {
  unsigned long long ga = (unsigned long long)(const void*)gsrc;
  u32x4 g0;
  g0[0] = 1u;                                         // count=1 (valid user D#)
  g0[1] = lds_addr;                                   // lds_addr (bytes)
  g0[2] = (unsigned)(ga & 0xFFFFFFFFu);               // global_addr[31:0]
  g0[3] = (unsigned)((ga >> 32) & 0x01FFFFFFu)        // global_addr[56:32]
        | (2u << 30);                                 // type=2 ("image")
  i32x8 g1;
  g1[0] = (1 << 16)                                   // data_size=1 -> 2 bytes
        | (1 << 20)                                   // pad_enable
        | (3 << 22)                                   // pad_interval: 16 DWORDs (64B)
        | (1 << 25);                                  // pad_amount: 2 DWORDs (8B)
  g1[1] = (int)((tensor_k & 0xFFFFu) << 16);          // tensor_dim0[15:0]
  g1[2] = (int)((tensor_k >> 16) & 0xFFFFu)           // tensor_dim0[31:16]
        | (int)((tensor_rows & 0xFFFFu) << 16);       // tensor_dim1[15:0]
  g1[3] = (int)((tensor_rows >> 16) & 0xFFFFu)        // tensor_dim1[31:16]
        | (int)((tile_k & 0xFFFFu) << 16);            // tile_dim0
  g1[4] = (int)(tile_rows & 0xFFFFu);                 // tile_dim1 (tile_dim2=0)
  g1[5] = (int)stride_elems;                          // tensor_dim0_stride[31:0]
  g1[6] = 0;                                          // stride hi / dim1_stride lo
  g1[7] = 0;
  i32x4 z4 = {0, 0, 0, 0};
#if __clang_major__ >= 23
  i32x8 z8 = {0, 0, 0, 0, 0, 0, 0, 0};
  __builtin_amdgcn_tensor_load_to_lds(g0, g1, z4, z4, z8, 0);
#else
  __builtin_amdgcn_tensor_load_to_lds(g0, g1, z4, z4, 0);
#endif
}
#endif  // HAVE_TDM

// ---------------- fused residual-add + LayerNorm ----------------
// out / out_bf each nullable: layer LNs write only the bf16 activation copy
// (GEMM consumes bf16), final LNs write f32 for the concat stage.
__global__ __launch_bounds__(256)
void add_ln_kernel(const float* __restrict__ hid, const float* __restrict__ res_in,
                   float* __restrict__ res_out, float* __restrict__ out,
                   bf16_t* __restrict__ out_bf,
                   const float* __restrict__ w, const float* __restrict__ b) {
  __shared__ float red[256];
  const int row = blockIdx.x, tid = threadIdx.x;
  const size_t base = (size_t)row * D;
  float v[4];
  float s = 0.f;
#pragma unroll
  for (int j = 0; j < 4; ++j) {
    int c = tid + j * 256;
    float t = hid[base + c];
    if (res_in) t += res_in[base + c];
    v[j] = t;
    s += t;
  }
  float mu = block_reduce_sum(s, red, tid) * (1.f / D);
  float vs = 0.f;
#pragma unroll
  for (int j = 0; j < 4; ++j) { float d0 = v[j] - mu; vs += d0 * d0; }
  float var  = block_reduce_sum(vs, red, tid) * (1.f / D);
  float rstd = rsqrtf(var + 1e-5f);
#pragma unroll
  for (int j = 0; j < 4; ++j) {
    int c = tid + j * 256;
    if (res_out) res_out[base + c] = v[j];
    float o = (v[j] - mu) * rstd * w[c] + b[c];
    if (out)    out[base + c]    = o;
    if (out_bf) out_bf[base + c] = (bf16_t)o;
  }
}

// ---------------- f32 -> bf16 weight conversion (once per launch) ----------------
__global__ __launch_bounds__(256)
void cvt_bf16_kernel(const float* __restrict__ src, bf16_t* __restrict__ dst, int n) {
  int i = (blockIdx.x * 256 + threadIdx.x) * 4;
  if (i + 3 < n) {
#pragma unroll
    for (int j = 0; j < 4; ++j) dst[i + j] = (bf16_t)src[i + j];
  }
}

// ======================================================================
// WMMA fragment helpers (CDNA5 wave32 16-bit layouts, ISA 7.12.2)
// ======================================================================
template <int LDW>
__device__ __forceinline__ v16bf load_afrag(const bf16_t (*As)[LDW], int mrow, int half) {
  v16bf f{};
#pragma unroll
  for (int e = 0; e < 16; ++e) {
    int p = e >> 1, wsel = e & 1;
    int ka = (p < 4) ? (half * 8 + p * 2 + wsel)
                     : (16 + half * 8 + (p - 4) * 2 + wsel);
    f[e] = As[mrow][ka];
  }
  return f;
}
template <int LDW>
__device__ __forceinline__ v16bf load_bfrag(const bf16_t (*Ws)[LDW], int ncol, int half) {
  v16bf f{};
#pragma unroll
  for (int e = 0; e < 16; ++e) f[e] = Ws[ncol][(e < 8) ? (half * 8 + e) : (16 + half * 8 + (e - 8))];
  return f;
}

// ---------------- big WMMA GEMM: BM=128, BN=64, 32x32 per wave ----------------
// bf16 operands from global. TDM path: wave0 issues two tensor_load_to_lds per
// K-step into the ping-pong buffer, DMA overlaps the 4 WMMAs; publish with
// s_wait_tensorcnt 0 + workgroup barrier. Fallback: synchronous bf16 staging.
// EPI: 0 none, 1 +bias, 2 softplus(+bias)
template <int EPI>
__global__ __launch_bounds__(256)
void gemm_wmma_big_kernel(const bf16_t* __restrict__ A, const bf16_t* __restrict__ W,
                          const float* __restrict__ bias, float* __restrict__ C,
                          int M, int N, int Kd, int lda) {
  constexpr int BM = 128, BN = 64, BK = 32, LDW = BK + 4;
#if HAVE_TDM
  __shared__ bf16_t As[2][BM][LDW];
  __shared__ bf16_t Ws[2][BN][LDW];
#else
  __shared__ bf16_t As[1][BM][LDW];
  __shared__ bf16_t Ws[1][BN][LDW];
#endif
  const int tid  = threadIdx.x;
  const int lane = tid & 31;
  const int wave = tid >> 5;          // 8 waves: 4(M) x 2(N) tiles of 32x32
  const int wm   = wave & 3;
  const int wn   = wave >> 2;
  const int row0 = blockIdx.y * BM;
  const int col0 = blockIdx.x * BN;

  const int half = lane >> 4;
  const int ml0  = wm * 32 + (lane & 15);
  const int nl0  = wn * 32 + (lane & 15);

  v8f acc00 = {}, acc01 = {}, acc10 = {}, acc11 = {};

  auto compute = [&](int cb) {
    v16bf a0 = load_afrag<LDW>(As[cb], ml0,      half);
    v16bf a1 = load_afrag<LDW>(As[cb], ml0 + 16, half);
    v16bf b0 = load_bfrag<LDW>(Ws[cb], nl0,      half);
    v16bf b1 = load_bfrag<LDW>(Ws[cb], nl0 + 16, half);
    acc00 = __builtin_amdgcn_wmma_f32_16x16x32_bf16(false, a0, false, b0, (short)0, acc00, false, false);
    acc01 = __builtin_amdgcn_wmma_f32_16x16x32_bf16(false, a0, false, b1, (short)0, acc01, false, false);
    acc10 = __builtin_amdgcn_wmma_f32_16x16x32_bf16(false, a1, false, b0, (short)0, acc10, false, false);
    acc11 = __builtin_amdgcn_wmma_f32_16x16x32_bf16(false, a1, false, b1, (short)0, acc11, false, false);
  };

#if HAVE_TDM
  auto issue = [&](int buf, int k0) {
    tdm_load_tile2d((unsigned)(unsigned long long)(&As[buf][0][0]),
                    A + (size_t)row0 * lda + k0, BK, BM,
                    (unsigned)Kd, (unsigned)M, (unsigned)lda);
    tdm_load_tile2d((unsigned)(unsigned long long)(&Ws[buf][0][0]),
                    W + (size_t)col0 * Kd + k0, BK, BN,
                    (unsigned)Kd, (unsigned)N, (unsigned)Kd);
  };
  const int nsteps = Kd / BK;
  if (tid == 0) {                 // one wave drives the DMA; EXEC is ignored by TDM
    issue(0, 0);
    __builtin_amdgcn_s_wait_tensorcnt(0);
  }
  __syncthreads();
  for (int ks = 0; ks < nsteps; ++ks) {
    const int cb = ks & 1;
    if (tid == 0 && ks + 1 < nsteps) issue(cb ^ 1, (ks + 1) * BK);  // DMA next tile
    compute(cb);                                                     // WMMA current
    if (tid == 0) __builtin_amdgcn_s_wait_tensorcnt(0);
    __syncthreads();
  }
#else
  const int ar = tid >> 1, ac = (tid & 1) * 16;   // 16 bf16 per thread (A)
  const int wr = tid >> 2, wc = (tid & 3) * 8;    // 8 bf16 per thread (W)
  for (int k0 = 0; k0 < Kd; k0 += BK) {
    {
      const bf16_t* src = A + (size_t)(row0 + ar) * lda + k0 + ac;
#pragma unroll
      for (int j = 0; j < 16; ++j) As[0][ar][ac + j] = src[j];
    }
    {
      const bf16_t* src = W + (size_t)(col0 + wr) * Kd + k0 + wc;
#pragma unroll
      for (int j = 0; j < 8; ++j) Ws[0][wr][wc + j] = src[j];
    }
    __syncthreads();
    if (k0 + BK < Kd) {
      __builtin_prefetch(A + (size_t)(row0 + ar) * lda + k0 + BK + ac, 0, 1);
      __builtin_prefetch(W + (size_t)(col0 + wr) * Kd + k0 + BK + wc, 0, 1);
    }
    compute(0);
    __syncthreads();
  }
#endif

  const int nA = col0 + wn * 32 + (lane & 15);
  const int nB = nA + 16;
  float bvA = 0.f, bvB = 0.f;
  if (EPI >= 1) { bvA = bias[nA]; bvB = bias[nB]; }
#pragma unroll
  for (int r = 0; r < 8; ++r) {
    int mA = row0 + wm * 32 + r + 8 * half;
    int mB = mA + 16;
    float v00 = acc00[r], v01 = acc01[r], v10 = acc10[r], v11 = acc11[r];
    if (EPI >= 1) { v00 += bvA; v01 += bvB; v10 += bvA; v11 += bvB; }
    if (EPI == 2) {
      v00 = softplusf(v00); v01 = softplusf(v01);
      v10 = softplusf(v10); v11 = softplusf(v11);
    }
    C[(size_t)mA * N + nA] = v00;
    C[(size_t)mA * N + nB] = v01;
    C[(size_t)mB * N + nA] = v10;
    C[(size_t)mB * N + nB] = v11;
  }
}

// ---------------- narrow WMMA GEMM (BN=32) for N=96 (xproj) ----------------
// Writes f32 C (scan consumes B/C) and optional bf16 copy C2 (dt GEMM input).
__global__ __launch_bounds__(256)
void gemm_wmma_narrow_kernel(const bf16_t* __restrict__ A, const bf16_t* __restrict__ W,
                             float* __restrict__ C, bf16_t* __restrict__ C2,
                             int M, int N, int Kd, int lda) {
  constexpr int BM = 64, BN = 32, BK = 32, LDW = BK + 4;
  __shared__ bf16_t As[BM][LDW];
  __shared__ bf16_t Ws[BN][LDW];
  (void)M;

  const int tid  = threadIdx.x;
  const int lane = tid & 31;
  const int wave = tid >> 5;
  const int wm   = wave & 3;
  const int wn   = wave >> 2;
  const int row0 = blockIdx.y * BM;
  const int col0 = blockIdx.x * BN;

  const int ar = tid >> 2, ac = (tid & 3) * 8;
  const int wr = tid >> 3, wc = (tid & 7) * 4;

  v8f acc = {};
  const int half = lane >> 4;
  const int ml   = wm * 16 + (lane & 15);
  const int nl   = wn * 16 + (lane & 15);

  for (int k0 = 0; k0 < Kd; k0 += BK) {
    {
      const bf16_t* src = A + (size_t)(row0 + ar) * lda + k0 + ac;
#pragma unroll
      for (int j = 0; j < 8; ++j) As[ar][ac + j] = src[j];
    }
    {
      const bf16_t* src = W + (size_t)(col0 + wr) * Kd + k0 + wc;
#pragma unroll
      for (int j = 0; j < 4; ++j) Ws[wr][wc + j] = src[j];
    }
    __syncthreads();
    if (k0 + BK < Kd) {
      __builtin_prefetch(A + (size_t)(row0 + ar) * lda + k0 + BK + ac, 0, 1);
      __builtin_prefetch(W + (size_t)(col0 + wr) * Kd + k0 + BK + wc, 0, 1);
    }
    v16bf af = load_afrag<LDW>(As, ml, half);
    v16bf bf = load_bfrag<LDW>(Ws, nl, half);
    acc = __builtin_amdgcn_wmma_f32_16x16x32_bf16(false, af, false, bf, (short)0, acc, false, false);
    __syncthreads();
  }

  const int n = col0 + wn * 16 + (lane & 15);
#pragma unroll
  for (int r = 0; r < 8; ++r) {
    int m = row0 + wm * 16 + r + 8 * half;
    float v = acc[r];
    C[(size_t)m * N + n] = v;
    if (C2) C2[(size_t)m * N + n] = (bf16_t)v;
  }
}

// ---------------- depthwise causal conv1d (K=4) + SiLU ----------------
__global__ __launch_bounds__(256)
void conv_silu_kernel(const float* __restrict__ xz, const float* __restrict__ cw,
                      const float* __restrict__ cb, float* __restrict__ u,
                      bf16_t* __restrict__ u_bf) {
  const int idx = blockIdx.x * 256 + threadIdx.x;   // over NTOK*DIN
  const int d   = idx & (DIN - 1);
  const int row = idx >> 11;
  const int t   = row & (L - 1);
  const int bb  = row >> 11;
  float acc = cb[d];
  const float* w = cw + d * KC;
#pragma unroll
  for (int j = 0; j < KC; ++j) {
    int tt = t - (KC - 1) + j;
    if (tt >= 0)
      acc += w[j] * xz[(size_t)(bb * L + tt) * (2 * DIN) + d];
  }
  float val = siluf(acc);
  u[idx]    = val;            // f32 for the scan
  u_bf[idx] = (bf16_t)val;    // bf16 for xproj GEMM
}

// ======================================================================
// Chunked selective scan (3 passes; chunk transfer = exp(A * sum dt))
// layouts: hloc/hin: [bb][c][s][d] ; dtsum: [bb][c][d]
// ======================================================================
__global__ __launch_bounds__(256)
void scan_part1_kernel(const float* __restrict__ dt, const float* __restrict__ xdbl,
                       const float* __restrict__ u, const float* __restrict__ Alog,
                       float* __restrict__ hloc, float* __restrict__ dtsum) {
  const int gid = blockIdx.x * 256 + threadIdx.x;   // [BATCH][CH][DIN]
  const int d   = gid & (DIN - 1);
  const int c   = (gid >> 11) & (CH - 1);
  const int bb  = gid >> 17;
  float Ar[S], h[S];
#pragma unroll
  for (int s = 0; s < S; ++s) { Ar[s] = -__expf(Alog[d * S + s]); h[s] = 0.f; }
  float ds = 0.f;
  for (int tt = 0; tt < CLEN; ++tt) {
    const size_t row = (size_t)bb * L + c * CLEN + tt;
    const float dtv = dt[row * DIN + d];
    const float du  = dtv * u[row * DIN + d];
    const float* xb = xdbl + row * XDBLW;
    ds += dtv;
#pragma unroll
    for (int s = 0; s < S; ++s)
      h[s] = __expf(dtv * Ar[s]) * h[s] + du * xb[R + s];
  }
  dtsum[(size_t)(bb * CH + c) * DIN + d] = ds;
#pragma unroll
  for (int s = 0; s < S; ++s)
    hloc[((size_t)(bb * CH + c) * S + s) * DIN + d] = h[s];
}

__global__ __launch_bounds__(256)
void scan_part2_kernel(const float* __restrict__ Alog, const float* __restrict__ dtsum,
                       const float* __restrict__ hloc, float* __restrict__ hin) {
  const int gid = blockIdx.x * 256 + threadIdx.x;   // [BATCH][DIN]
  const int d   = gid & (DIN - 1);
  const int bb  = gid >> 11;
  float Ar[S], h[S];
#pragma unroll
  for (int s = 0; s < S; ++s) { Ar[s] = -__expf(Alog[d * S + s]); h[s] = 0.f; }
  for (int c = 0; c < CH; ++c) {
    const size_t base = (size_t)(bb * CH + c);
#pragma unroll
    for (int s = 0; s < S; ++s)
      hin[(base * S + s) * DIN + d] = h[s];
    const float ds = dtsum[base * DIN + d];
#pragma unroll
    for (int s = 0; s < S; ++s)
      h[s] = __expf(ds * Ar[s]) * h[s] + hloc[(base * S + s) * DIN + d];
  }
}

__global__ __launch_bounds__(256)
void scan_part3_kernel(const float* __restrict__ dt, const float* __restrict__ xdbl,
                       const float* __restrict__ u, const float* __restrict__ Alog,
                       const float* __restrict__ hin, float* __restrict__ y) {
  const int gid = blockIdx.x * 256 + threadIdx.x;   // [BATCH][CH][DIN]
  const int d   = gid & (DIN - 1);
  const int c   = (gid >> 11) & (CH - 1);
  const int bb  = gid >> 17;
  float Ar[S], h[S];
#pragma unroll
  for (int s = 0; s < S; ++s) {
    Ar[s] = -__expf(Alog[d * S + s]);
    h[s]  = hin[((size_t)(bb * CH + c) * S + s) * DIN + d];
  }
  for (int tt = 0; tt < CLEN; ++tt) {
    const size_t row = (size_t)bb * L + c * CLEN + tt;
    const float dtv = dt[row * DIN + d];
    const float du  = dtv * u[row * DIN + d];
    const float* xb = xdbl + row * XDBLW;
    float acc = 0.f;
#pragma unroll
    for (int s = 0; s < S; ++s) {
      h[s] = __expf(dtv * Ar[s]) * h[s] + du * xb[R + s];
      acc += h[s] * xb[R + S + s];
    }
    y[row * DIN + d] = acc;
  }
}

// ---------------- gate: y_bf = bf16((y + u*Dp) * silu(z)) ----------------
__global__ __launch_bounds__(256)
void gate_kernel(const float* __restrict__ y, const float* __restrict__ u,
                 const float* __restrict__ xz, const float* __restrict__ Dp,
                 bf16_t* __restrict__ y_bf) {
  const int idx = blockIdx.x * 256 + threadIdx.x;
  const int d   = idx & (DIN - 1);
  const int row = idx >> 11;
  const float z = xz[(size_t)row * (2 * DIN) + DIN + d];
  y_bf[idx] = (bf16_t)((y[idx] + u[idx] * Dp[d]) * siluf(z));
}

// ---------------- sequence flip / concat-with-flip ----------------
__global__ __launch_bounds__(256)
void flip_rows_kernel(const float* __restrict__ src, float* __restrict__ dst) {
  const int row = blockIdx.x, tid = threadIdx.x;
  const int t = row & (L - 1), bb = row >> 11;
  const int srow = bb * L + (L - 1 - t);
  for (int c = tid; c < D; c += 256)
    dst[(size_t)row * D + c] = src[(size_t)srow * D + c];
}

__global__ __launch_bounds__(256)
void concat_flip_kernel(const float* __restrict__ fh, const float* __restrict__ bh,
                        bf16_t* __restrict__ merged) {
  const int row = blockIdx.x, tid = threadIdx.x;
  const int t = row & (L - 1), bb = row >> 11;
  const int brow = bb * L + (L - 1 - t);
  for (int c = tid; c < D; c += 256) {
    merged[(size_t)row * (2 * D) + c]     = (bf16_t)fh[(size_t)row * D + c];
    merged[(size_t)row * (2 * D) + D + c] = (bf16_t)bh[(size_t)brow * D + c];
  }
}

// ---------------- host orchestration ----------------
struct DirParams {
  const float *nw, *nb, *inW, *cw, *cb, *xpW, *dtW, *dtb, *Alog, *Dp, *outW;
};
struct DirParamsBf {        // bf16 weight copies in workspace
  bf16_t *inW, *xpW, *dtW, *outW;
};

extern "C" void kernel_launch(void* const* d_in, const int* in_sizes, int n_in,
                              void* d_out, int out_size, void* d_ws, size_t ws_size,
                              hipStream_t stream) {
  (void)in_sizes; (void)n_in; (void)out_size; (void)ws_size;

  const float* x = (const float*)d_in[0];
  auto dir = [&](int base) {
    DirParams p;
    p.nw   = (const float*)d_in[base + 0];
    p.nb   = (const float*)d_in[base + 1];
    p.inW  = (const float*)d_in[base + 2];
    p.cw   = (const float*)d_in[base + 3];
    p.cb   = (const float*)d_in[base + 4];
    p.xpW  = (const float*)d_in[base + 5];
    p.dtW  = (const float*)d_in[base + 6];
    p.dtb  = (const float*)d_in[base + 7];
    p.Alog = (const float*)d_in[base + 8];
    p.Dp   = (const float*)d_in[base + 9];
    p.outW = (const float*)d_in[base + 10];
    return p;
  };
  const DirParams fw = dir(1), bw = dir(12);
  const float* normf_w = (const float*)d_in[23];
  const float* normf_b = (const float*)d_in[24];
  const float* merge_W = (const float*)d_in[25];
  const float* merge_b = (const float*)d_in[26];

  // -------- workspace carve (bump allocator, ~460 MB total) --------
  char* wsp = (char*)d_ws;
  size_t off = 0;
  auto carve = [&](size_t elems) -> float* {
    float* p = (float*)(wsp + off);
    off += (elems * sizeof(float) + 255) & ~(size_t)255;
    return p;
  };
  auto carve_bf = [&](size_t elems) -> bf16_t* {
    bf16_t* p = (bf16_t*)(wsp + off);
    off += (elems * sizeof(bf16_t) + 255) & ~(size_t)255;
    return p;
  };
  float* xrev     = carve((size_t)NTOK * D);
  float* residual = carve((size_t)NTOK * D);
  float* hidden   = carve((size_t)NTOK * D);
  float* xz       = carve((size_t)NTOK * 2 * DIN);
  float* u        = carve((size_t)NTOK * DIN);
  float* xdbl     = carve((size_t)NTOK * XDBLW);
  float* dt       = carve((size_t)NTOK * DIN);
  float* yscan    = carve((size_t)NTOK * DIN);
  float* hloc     = carve((size_t)BATCH * CH * S * DIN);
  float* hin      = carve((size_t)BATCH * CH * S * DIN);
  float* dtsum    = carve((size_t)BATCH * CH * DIN);
  float* fh_ln    = carve((size_t)NTOK * D);
  float* bh_ln    = carve((size_t)NTOK * D);

  bf16_t* hn_bf     = carve_bf((size_t)NTOK * D);
  bf16_t* u_bf      = carve_bf((size_t)NTOK * DIN);
  bf16_t* xdbl_bf   = carve_bf((size_t)NTOK * XDBLW);
  bf16_t* y_bf      = carve_bf((size_t)NTOK * DIN);
  bf16_t* merged_bf = carve_bf((size_t)NTOK * 2 * D);

  DirParamsBf fwb, bwb;
  bf16_t* mergeW_bf;
  auto carve_dir_bf = [&](DirParamsBf& q) {
    q.inW  = carve_bf((size_t)NL * 2 * DIN * D);
    q.xpW  = carve_bf((size_t)NL * XDBLW * DIN);
    q.dtW  = carve_bf((size_t)NL * DIN * R);
    q.outW = carve_bf((size_t)NL * D * DIN);
  };
  carve_dir_bf(fwb);
  carve_dir_bf(bwb);
  mergeW_bf = carve_bf((size_t)D * 2 * D);

  const dim3 blk(256);
  const int  ew_grid    = (NTOK * DIN) / 256;        // elementwise [NTOK, DIN]
  const int  chunk_grid = (BATCH * CH * DIN) / 256;  // scan passes 1 & 3

  // -------- one-time (per launch) weight conversion to bf16 --------
  auto cvt = [&](const float* s, bf16_t* d0, size_t n) {
    cvt_bf16_kernel<<<(unsigned)((n + 1023) / 1024), blk, 0, stream>>>(s, d0, (int)n);
  };
  cvt(fw.inW,  fwb.inW,  (size_t)NL * 2 * DIN * D);
  cvt(fw.xpW,  fwb.xpW,  (size_t)NL * XDBLW * DIN);
  cvt(fw.dtW,  fwb.dtW,  (size_t)NL * DIN * R);
  cvt(fw.outW, fwb.outW, (size_t)NL * D * DIN);
  cvt(bw.inW,  bwb.inW,  (size_t)NL * 2 * DIN * D);
  cvt(bw.xpW,  bwb.xpW,  (size_t)NL * XDBLW * DIN);
  cvt(bw.dtW,  bwb.dtW,  (size_t)NL * DIN * R);
  cvt(bw.outW, bwb.outW, (size_t)NL * D * DIN);
  cvt(merge_W, mergeW_bf, (size_t)D * 2 * D);

  flip_rows_kernel<<<NTOK, blk, 0, stream>>>(x, xrev);

  auto run_stack = [&](const float* xin, const DirParams& P, const DirParamsBf& Q,
                       float* out_ln) {
    const float* hid = xin;
    for (int i = 0; i < NL; ++i) {
      // residual += hidden ; hn_bf = bf16(LN(residual))
      add_ln_kernel<<<NTOK, blk, 0, stream>>>(
          hid, (i == 0) ? nullptr : residual, residual, nullptr, hn_bf,
          P.nw + (size_t)i * D, P.nb + (size_t)i * D);
      // in_proj: [NTOK,D] @ [4096,D]^T -> xz (f32)
      gemm_wmma_big_kernel<0><<<dim3((2 * DIN) / 64, NTOK / 128), blk, 0, stream>>>(
          hn_bf, Q.inW + (size_t)i * 2 * DIN * D, nullptr, xz, NTOK, 2 * DIN, D, D);
      // depthwise causal conv + SiLU -> u (f32) + u_bf
      conv_silu_kernel<<<ew_grid, blk, 0, stream>>>(
          xz, P.cw + (size_t)i * DIN * KC, P.cb + (size_t)i * DIN, u, u_bf);
      // x_proj: [NTOK,DIN] @ [96,DIN]^T -> xdbl (f32) + xdbl_bf
      gemm_wmma_narrow_kernel<<<dim3(XDBLW / 32, NTOK / 64), blk, 0, stream>>>(
          u_bf, Q.xpW + (size_t)i * XDBLW * DIN, xdbl, xdbl_bf, NTOK, XDBLW, DIN, DIN);
      // dt = softplus(dtr @ dtW^T + dtb)   (A = first R cols of xdbl_bf, lda=96)
      gemm_wmma_big_kernel<2><<<dim3(DIN / 64, NTOK / 128), blk, 0, stream>>>(
          xdbl_bf, Q.dtW + (size_t)i * DIN * R, P.dtb + (size_t)i * DIN, dt,
          NTOK, DIN, R, XDBLW);
      // chunked selective scan
      scan_part1_kernel<<<chunk_grid, blk, 0, stream>>>(
          dt, xdbl, u, P.Alog + (size_t)i * DIN * S, hloc, dtsum);
      scan_part2_kernel<<<(BATCH * DIN) / 256, blk, 0, stream>>>(
          P.Alog + (size_t)i * DIN * S, dtsum, hloc, hin);
      scan_part3_kernel<<<chunk_grid, blk, 0, stream>>>(
          dt, xdbl, u, P.Alog + (size_t)i * DIN * S, hin, yscan);
      // y_bf = bf16((y + u*Dp) * silu(z))
      gate_kernel<<<ew_grid, blk, 0, stream>>>(
          yscan, u, xz, P.Dp + (size_t)i * DIN, y_bf);
      // out_proj: [NTOK,DIN] @ [D,DIN]^T -> hidden (f32)
      gemm_wmma_big_kernel<0><<<dim3(D / 64, NTOK / 128), blk, 0, stream>>>(
          y_bf, Q.outW + (size_t)i * D * DIN, nullptr, hidden, NTOK, D, DIN, DIN);
      hid = hidden;
    }
    add_ln_kernel<<<NTOK, blk, 0, stream>>>(
        hidden, residual, nullptr, out_ln, nullptr, normf_w, normf_b);
  };

  run_stack(x,    fw, fwb, fh_ln);
  run_stack(xrev, bw, bwb, bh_ln);

  // merged_bf = bf16([fh_ln, flip(bh_ln)]) ; out = merged @ merge_W^T + merge_b
  concat_flip_kernel<<<NTOK, blk, 0, stream>>>(fh_ln, bh_ln, merged_bf);
  gemm_wmma_big_kernel<1><<<dim3(D / 64, NTOK / 128), blk, 0, stream>>>(
      merged_bf, mergeW_bf, merge_b, (float*)d_out, NTOK, D, 2 * D, 2 * D);
}
